// MultiHeadAttention_56461640073327
// MI455X (gfx1250) — matrix operand order repaired and checked by reading, not compiled, not run on hardware
//
#include <hip/hip_runtime.h>
#include <hip/hip_bf16.h>
#include <stdint.h>

// ---------------------------------------------------------------------------
// MultiHeadAttention forward for MI455X (gfx1250, wave32, WMMA + TDM).
// B=2, T=2048, DIM=1024, NH=16, HD=64.
// ---------------------------------------------------------------------------

typedef __attribute__((ext_vector_type(16))) _Float16 v16h;
typedef __attribute__((ext_vector_type(8)))  _Float16 v8h;
typedef __attribute__((ext_vector_type(4)))  _Float16 v4h;
typedef __attribute__((ext_vector_type(8)))  float    v8f;
typedef __attribute__((ext_vector_type(4)))  unsigned int u32x4;
typedef __attribute__((ext_vector_type(8)))  int      i32x8;
typedef __attribute__((ext_vector_type(4)))  int      i32x4;

constexpr int Bc   = 2;
constexpr int Tc   = 2048;
constexpr int DIMc = 1024;
constexpr int NHc  = 16;
constexpr int HDc  = 64;
constexpr int BTc  = Bc * Tc;  // 4096

#if __has_builtin(__builtin_amdgcn_tensor_load_to_lds)
#define USE_TDM 1
#else
#define USE_TDM 0
#endif

// ---------------------------------------------------------------------------
// WMMA fragment loaders, following CDNA5 ISA 7.12.2 layouts.
// A (16x32 f16): lane l -> row m=l%16, g=l/16.
//   half i (0..7)  : K = 8*g + i          ; half i (8..15): K = 16+8*g+(i-8)
// -> per lane two contiguous 16B chunks at [k0+8g] and [k0+16+8g].
// ---------------------------------------------------------------------------
__device__ __forceinline__ v16h load_a16x32(const _Float16* base, int ld,
                                            int m0, int k0, int lane) {
  const int m = lane & 15, g = lane >> 4;
  const _Float16* p = base + (size_t)(m0 + m) * ld + k0 + 8 * g;
  v8h lo = *(const v8h*)(p);
  v8h hi = *(const v8h*)(p + 16);
  v16h r;
#pragma unroll
  for (int i = 0; i < 8; ++i) { r[i] = lo[i]; r[i + 8] = hi[i]; }
  return r;
}

// B (32x16 f16): lane l -> col n=l%16, g=l/16; half i (0..7): K = 8*g + i ; half i (8..15): K = 16+8*g+(i-8).
// Source layout: B(k,n) = base[(n0+n)*ld + k0+k]  (K contiguous per "row").
__device__ __forceinline__ v16h load_b32x16(const _Float16* base, int ld,
                                            int n0, int k0, int lane) {
  const int n = lane & 15, g = lane >> 4;
  const _Float16* p = base + (size_t)(n0 + n) * ld + k0 + 8 * g;
  v8h lo = *(const v8h*)(p);
  v8h hi = *(const v8h*)(p + 16);
  v16h r;
#pragma unroll
  for (int i = 0; i < 8; ++i) { r[i] = lo[i]; r[i + 8] = hi[i]; }
  return r;
}

__device__ __forceinline__ v8f wmma_f16(v16h a, v16h b, v8f c) {
  return __builtin_amdgcn_wmma_f32_16x16x32_f16(false, a, false, b,
                                                (short)0, c, false, false);
}

// ---------------------------------------------------------------------------
// Tensor Data Mover: 2-D tile load, D# built per CDNA5 ISA.
// ---------------------------------------------------------------------------
__device__ __forceinline__ unsigned lds_off_of(const void* p) {
  return (unsigned)(uintptr_t)(__attribute__((address_space(3))) const void*)p;
}

#if USE_TDM
__device__ __forceinline__ void tdm_load_2d(const _Float16* gsrc,
                                            unsigned lds_off,
                                            unsigned tile_d0, unsigned tile_d1,
                                            unsigned tensor_d0,
                                            unsigned tensor_d1,
                                            unsigned stride0) {
  const uint64_t ga = (uint64_t)(uintptr_t)gsrc;
  u32x4 g0;
  g0[0] = 1u;                              // count=1, user descriptor
  g0[1] = lds_off;                         // lds_addr [63:32]
  g0[2] = (unsigned)ga;                    // global_addr [95:64]
  g0[3] = (unsigned)((ga >> 32) & 0x1FFFFFFu) | 0x80000000u;  // addr hi, type=2
  i32x8 g1;
  g1[0] = 0x00010000;                      // wg_mask=0, data_size=1 (2 bytes)
  g1[1] = (int)((tensor_d0 & 0xFFFFu) << 16);                 // dim0 lo16
  g1[2] = (int)((tensor_d0 >> 16) | ((tensor_d1 & 0xFFFFu) << 16));
  g1[3] = (int)((tensor_d1 >> 16) | (tile_d0 << 16));         // tile_dim0
  g1[4] = (int)(tile_d1 & 0xFFFFu);        // tile_dim1 (tile_dim2 = 0)
  g1[5] = (int)stride0;                    // tensor_dim0_stride lo32
  g1[6] = 0;                               // stride0 hi16 / stride1 lo16
  g1[7] = 0;                               // stride1 hi32
  const i32x4 z4 = {0, 0, 0, 0};
#if __clang_major__ >= 23
  const i32x8 z8 = {0, 0, 0, 0, 0, 0, 0, 0};
  __builtin_amdgcn_tensor_load_to_lds(g0, g1, z4, z4, z8, 0);
#else
  __builtin_amdgcn_tensor_load_to_lds(g0, g1, z4, z4, 0);
#endif
}
#endif

__device__ __forceinline__ void tensor_wait() {
#if __has_builtin(__builtin_amdgcn_s_wait_tensorcnt)
  __builtin_amdgcn_s_wait_tensorcnt(0);
#else
  asm volatile("s_wait_tensorcnt 0x0" ::: "memory");
#endif
}

// ---------------------------------------------------------------------------
// fp32 -> f16 conversion (vector x4)
// ---------------------------------------------------------------------------
__global__ void cvt_f32_f16(const float* __restrict__ in,
                            _Float16* __restrict__ out, int n) {
  int i = (blockIdx.x * blockDim.x + threadIdx.x) * 4;
  if (i < n) {
    float4 v = *(const float4*)(in + i);
    v4h h;
    h[0] = (_Float16)v.x; h[1] = (_Float16)v.y;
    h[2] = (_Float16)v.z; h[3] = (_Float16)v.w;
    *(v4h*)(out + i) = h;
  }
}

// ---------------------------------------------------------------------------
// GEMM: C[m,n] = sum_k A[m,k] * W[n,k] + bias[n]
// Block = 256 threads (8 waves), tile 128(M) x 128(N); wave = 64x32.
// ---------------------------------------------------------------------------
template <bool OUT_F16>
__global__ __launch_bounds__(256) void gemm_kernel(
    const _Float16* __restrict__ A, const _Float16* __restrict__ W,
    const float* __restrict__ bias, void* __restrict__ Cv,
    int M, int N, int K) {
  const int lane = threadIdx.x & 31;
  const int wave = threadIdx.x >> 5;
  const int wm = wave >> 2, wn = wave & 3;
  const int m0 = blockIdx.y * 128 + wm * 64;
  const int n0 = blockIdx.x * 128 + wn * 32;
  const int n_l = lane & 15, g = lane >> 4;

  v8f c[4][2] = {};
  for (int k0 = 0; k0 < K; k0 += 32) {
    if (k0 + 64 < K) {  // global_prefetch_b8 of upcoming K-slab
      __builtin_prefetch(A + (size_t)(m0 + lane) * K + k0 + 64, 0, 0);
      __builtin_prefetch(W + (size_t)(n0 + lane) * K + k0 + 64, 0, 0);
    }
    v16h a[4], b[2];
#pragma unroll
    for (int i = 0; i < 4; ++i) a[i] = load_a16x32(A, K, m0 + 16 * i, k0, lane);
#pragma unroll
    for (int j = 0; j < 2; ++j) b[j] = load_b32x16(W, K, n0 + 16 * j, k0, lane);
#pragma unroll
    for (int i = 0; i < 4; ++i)
#pragma unroll
      for (int j = 0; j < 2; ++j) c[i][j] = wmma_f16(a[i], b[j], c[i][j]);
  }

#pragma unroll
  for (int j = 0; j < 2; ++j) {
    const int col = n0 + 16 * j + n_l;
    const float bb = bias[col];
#pragma unroll
    for (int i = 0; i < 4; ++i) {
#pragma unroll
      for (int r = 0; r < 8; ++r) {
        const int row = m0 + 16 * i + r + 8 * g;
        const float v = c[i][j][r] + bb;
        if (OUT_F16)
          ((_Float16*)Cv)[(size_t)row * N + col] = (_Float16)v;
        else
          ((float*)Cv)[(size_t)row * N + col] = v;
      }
    }
  }
}

// ---------------------------------------------------------------------------
// RoPE + layout shuffle (reproduces the reference's emb[::2]/emb[1::2] mix).
// q,k -> [B,NH,T,HD] f16 ; v -> transposed [B,NH,HD,T] f16.
// ---------------------------------------------------------------------------
__global__ void rope_kernel(const _Float16* __restrict__ qkv,
                            _Float16* __restrict__ q_r,
                            _Float16* __restrict__ k_r,
                            _Float16* __restrict__ v_t) {
  const int idx = blockIdx.x * blockDim.x + threadIdx.x;
  const int j = idx & 31;
  const int t = (idx >> 5) & (Tc - 1);
  const int h = (idx >> 16) & (NHc - 1);
  const int b = idx >> 20;
  if (b >= Bc) return;

  const _Float16* row = qkv + (size_t)(b * Tc + t) * (3 * DIMc);
  const _Float16* qp = row + h * HDc;
  const _Float16* kp = qp + DIMc;
  const _Float16* vp = qp + 2 * DIMc;

  const float C = 9.210340371976184f / 32.0f;  // ln(10000)/32
  const float tf = (float)t;
  float sj, cj;
  if (j < 16) {
    sj = __sinf(tf * __expf(-C * (float)(2 * j)));
    cj = __sinf(tf * __expf(-C * (float)(2 * j + 1)));
  } else {
    const int jj = j - 16;
    sj = __cosf(tf * __expf(-C * (float)(2 * jj)));
    cj = __cosf(tf * __expf(-C * (float)(2 * jj + 1)));
  }

  const float x1q = (float)qp[2 * j], x2q = (float)qp[2 * j + 1];
  const float x1k = (float)kp[2 * j], x2k = (float)kp[2 * j + 1];

  const size_t ob = (((size_t)b * NHc + h) * Tc + t) * HDc;
  q_r[ob + j]      = (_Float16)(x1q * cj - x2q * sj);
  q_r[ob + 32 + j] = (_Float16)(x1q * sj + x2q * cj);
  k_r[ob + j]      = (_Float16)(x1k * cj - x2k * sj);
  k_r[ob + 32 + j] = (_Float16)(x1k * sj + x2k * cj);

  const size_t vb = ((size_t)b * NHc + h) * HDc * Tc;
  v_t[vb + (size_t)(2 * j) * Tc + t]     = vp[2 * j];
  v_t[vb + (size_t)(2 * j + 1) * Tc + t] = vp[2 * j + 1];
}

// ---------------------------------------------------------------------------
// Flash attention (non-causal). Block = 128 threads = 4 waves; block covers
// 64 query rows (16/wave); grid = (T/64, B*NH).
// K/V tiles (32 keys) staged into LDS by the Tensor Data Mover, double
// buffered: wave0 issues the K-tile D#, wave1 the V-tile D#, issuers wait
// TENSORcnt, barrier publishes to the workgroup. Fallback: cooperative
// global->LDS copies when the TDM builtin is unavailable.
// ---------------------------------------------------------------------------
__global__ __launch_bounds__(128) void attn_kernel(
    const _Float16* __restrict__ q_r, const _Float16* __restrict__ k_r,
    const _Float16* __restrict__ v_t, _Float16* __restrict__ o) {
  __shared__ __align__(16) _Float16 ksh[2][32][64];   // [key][hd]   8 KB
  __shared__ __align__(16) _Float16 vsh[2][64][32];   // [hd][key]   8 KB
  __shared__ __align__(16) _Float16 pbuf[4][16][32];  // per-wave P  4 KB

  const int tid = threadIdx.x;
  const int lane = tid & 31;
  const int wave = tid >> 5;
  const int waveu = __builtin_amdgcn_readfirstlane(tid) >> 5;  // scalar wave id
  const int n_l = lane & 15, g = lane >> 4;
  const int bh = blockIdx.y;
  const int b = bh >> 4, h = bh & 15;
  const int q0 = blockIdx.x * 64 + wave * 16;

  const _Float16* qb = q_r + (size_t)bh * Tc * HDc;
  const _Float16* kb = k_r + (size_t)bh * Tc * HDc;
  const _Float16* vb = v_t + (size_t)bh * HDc * Tc;

  auto stage = [&](int buf, int key0) {
#if USE_TDM
    if (waveu == 0) {
      tdm_load_2d(kb + (size_t)key0 * HDc, lds_off_of(&ksh[buf][0][0]),
                  /*tile_d0=*/HDc, /*tile_d1=*/32,
                  /*tensor_d0=*/HDc, /*tensor_d1=*/Tc, /*stride0=*/HDc);
    } else if (waveu == 1) {
      tdm_load_2d(vb + key0, lds_off_of(&vsh[buf][0][0]),
                  /*tile_d0=*/32, /*tile_d1=*/HDc,
                  /*tensor_d0=*/Tc, /*tensor_d1=*/HDc, /*stride0=*/Tc);
    }
#else
#pragma unroll
    for (int c = tid; c < 256; c += 128) {
      { const int row = c >> 3, off = (c & 7) * 8;
        *(v8h*)&ksh[buf][row][off] =
            *(const v8h*)(kb + (size_t)(key0 + row) * HDc + off); }
      { const int row = c >> 2, off = (c & 3) * 8;
        *(v8h*)&vsh[buf][row][off] =
            *(const v8h*)(vb + (size_t)row * Tc + key0 + off); }
    }
#endif
  };
  auto stage_wait = [&]() {
#if USE_TDM
    if (waveu < 2) tensor_wait();
#endif
  };

  const v16h qa0 = load_a16x32(qb, HDc, q0, 0, lane);
  const v16h qa1 = load_a16x32(qb, HDc, q0, 32, lane);

  float mrun[8], lrun[8];
  v8f acc[4] = {};
#pragma unroll
  for (int r = 0; r < 8; ++r) { mrun[r] = -1e30f; lrun[r] = 0.f; }

  int buf = 0;
  stage(0, 0);
  stage_wait();
  __syncthreads();

  for (int key0 = 0; key0 < Tc; key0 += 32) {
    const int nbuf = buf ^ 1;
    if (key0 + 32 < Tc) stage(nbuf, key0 + 32);  // TDM overlaps compute

    // S = Q K^T from LDS K tile (K-dim = HD = 64 -> 2 wmma / 16-key frag)
    const _Float16* kl = &ksh[buf][0][0];
    v8f s0 = {}, s1 = {};
    {
      v16h b00 = load_b32x16(kl, HDc, 0, 0, lane);
      v16h b01 = load_b32x16(kl, HDc, 0, 32, lane);
      v16h b10 = load_b32x16(kl, HDc, 16, 0, lane);
      v16h b11 = load_b32x16(kl, HDc, 16, 32, lane);
      s0 = wmma_f16(qa0, b00, s0);
      s0 = wmma_f16(qa1, b01, s0);
      s1 = wmma_f16(qa0, b10, s1);
      s1 = wmma_f16(qa1, b11, s1);
    }

    float p0[8], p1[8];
#pragma unroll
    for (int r = 0; r < 8; ++r) {
      s0[r] *= 0.125f;  // 1/sqrt(64)
      s1[r] *= 0.125f;
      float v = fmaxf(s0[r], s1[r]);
      v = fmaxf(v, __shfl_xor(v, 1, 32));
      v = fmaxf(v, __shfl_xor(v, 2, 32));
      v = fmaxf(v, __shfl_xor(v, 4, 32));
      v = fmaxf(v, __shfl_xor(v, 8, 32));  // row max within 16-lane half
      const float mnew = fmaxf(mrun[r], v);
      const float alpha = __expf(mrun[r] - mnew);
      mrun[r] = mnew;
      p0[r] = __expf(s0[r] - mnew);
      p1[r] = __expf(s1[r] - mnew);
      float rs = p0[r] + p1[r];
      rs += __shfl_xor(rs, 1, 32);
      rs += __shfl_xor(rs, 2, 32);
      rs += __shfl_xor(rs, 4, 32);
      rs += __shfl_xor(rs, 8, 32);
      lrun[r] = lrun[r] * alpha + rs;
#pragma unroll
      for (int jf = 0; jf < 4; ++jf) acc[jf][r] *= alpha;
    }

    // C-layout P -> per-wave LDS tile -> re-read as A fragment (in-order DS
    // within a wave keeps this safe without a barrier).
#pragma unroll
    for (int r = 0; r < 8; ++r) {
      pbuf[wave][r + 8 * g][n_l]      = (_Float16)p0[r];
      pbuf[wave][r + 8 * g][16 + n_l] = (_Float16)p1[r];
    }
    const v16h pa = load_a16x32(&pbuf[wave][0][0], 32, 0, 0, lane);

    // O += P V from LDS V tile
    const _Float16* vl = &vsh[buf][0][0];
#pragma unroll
    for (int jf = 0; jf < 4; ++jf) {
      v16h bv = load_b32x16(vl, 32, jf * 16, 0, lane);
      acc[jf] = wmma_f16(pa, bv, acc[jf]);
    }

    stage_wait();     // issuers drain TENSORcnt for the next buffer
    __syncthreads();  // publish next buffer / retire reads of current
    buf = nbuf;
  }

  // Normalize and store to [B*T, DIM] f16 (input of output projection).
#pragma unroll
  for (int r = 0; r < 8; ++r) {
    const float inv = 1.0f / lrun[r];
    const int t = q0 + r + 8 * g;
    const size_t base = (size_t)(b * Tc + t) * DIMc + h * HDc;
#pragma unroll
    for (int jf = 0; jf < 4; ++jf)
      o[base + jf * 16 + n_l] = (_Float16)(acc[jf][r] * inv);
  }
}

// ---------------------------------------------------------------------------
// Launch
// ---------------------------------------------------------------------------
extern "C" void kernel_launch(void* const* d_in, const int* in_sizes, int n_in,
                              void* d_out, int out_size, void* d_ws,
                              size_t ws_size, hipStream_t stream) {
  const float* x     = (const float*)d_in[0];
  const float* qkv_w = (const float*)d_in[1];
  const float* qkv_b = (const float*)d_in[2];
  const float* out_w = (const float*)d_in[3];
  const float* out_b = (const float*)d_in[4];
  float* out = (float*)d_out;

  char* ws = (char*)d_ws;
  size_t off = 0;
  auto take = [&](size_t bytes) -> char* {
    char* p = ws + off;
    off = (off + bytes + 255) & ~(size_t)255;
    return p;
  };
  _Float16* xh   = (_Float16*)take((size_t)BTc * DIMc * 2);       // x in f16
  _Float16* wh   = (_Float16*)take((size_t)3 * DIMc * DIMc * 2);  // qkv_w f16
  _Float16* owh  = (_Float16*)take((size_t)DIMc * DIMc * 2);      // out_w f16
  _Float16* q_r  = (_Float16*)take((size_t)BTc * DIMc * 2);       // rope'd Q
  _Float16* k_r  = (_Float16*)take((size_t)BTc * DIMc * 2);       // rope'd K
  _Float16* v_t  = (_Float16*)take((size_t)BTc * DIMc * 2);       // V^T
  _Float16* qkvr = (_Float16*)take((size_t)BTc * 3 * DIMc * 2);   // raw QKV
  _Float16* oacc = qkvr;  // attention output aliases dead qkv-raw region

  const int n1 = BTc * DIMc;
  const int n2 = 3 * DIMc * DIMc;
  const int n3 = DIMc * DIMc;
  cvt_f32_f16<<<n1 / 4 / 256, 256, 0, stream>>>(x, xh, n1);
  cvt_f32_f16<<<n2 / 4 / 256, 256, 0, stream>>>(qkv_w, wh, n2);
  cvt_f32_f16<<<n3 / 4 / 256, 256, 0, stream>>>(out_w, owh, n3);

  // QKV projection: [4096,1024] x [1024,3072]^T + b -> f16
  gemm_kernel<true><<<dim3(3 * DIMc / 128, BTc / 128), 256, 0, stream>>>(
      xh, wh, qkv_b, qkvr, BTc, 3 * DIMc, DIMc);

  // RoPE + head-major reshape (+ V transpose)
  rope_kernel<<<(Bc * NHc * Tc * 32) / 256, 256, 0, stream>>>(qkvr, q_r, k_r,
                                                              v_t);

  // Flash attention (TDM-staged K/V tiles)
  attn_kernel<<<dim3(Tc / 64, Bc * NHc), 128, 0, stream>>>(q_r, k_r, v_t,
                                                           oacc);

  // Output projection: [4096,1024] x [1024,1024]^T + b -> fp32 d_out
  gemm_kernel<false><<<dim3(DIMc / 128, BTc / 128), 256, 0, stream>>>(
      oacc, owh, out_b, out, BTc, DIMc, DIMc);
}
